// VF_1752346657348
// MI455X (gfx1250) — compile-verified, matrix operand order checked by reading, not run on hardware
//
#include <hip/hip_runtime.h>

// ---------------------------------------------------------------------------
// Problem constants (match reference)
// ---------------------------------------------------------------------------
#define NN  131072      // nodes
#define EE  1048576     // edges
#define CC  128         // channels
#define ACTD 16         // nodes per graph
#define BB  (NN / ACTD) // 8192 graphs

typedef __attribute__((ext_vector_type(16))) __bf16 v16bf;
typedef __attribute__((ext_vector_type(8)))  __bf16 bf16x8;
typedef __attribute__((ext_vector_type(8)))  float  v8f;

// Split an fp32 value into bf16 hi + bf16 lo (bf16x3 GEMM scheme).
__device__ __forceinline__ void bf_split(float f, __bf16& hi, __bf16& lo) {
    hi = (__bf16)f;
    lo = (__bf16)(f - (float)hi);
}

// ---------------------------------------------------------------------------
// Kernel 1: zero the accumulator region (agg[N*128] followed by deg[N])
// ---------------------------------------------------------------------------
__global__ void k_zero(float* __restrict__ p, size_t n) {
    size_t i = (size_t)blockIdx.x * blockDim.x + threadIdx.x;
    size_t stride = (size_t)gridDim.x * blockDim.x;
    for (; i < n; i += stride) p[i] = 0.0f;
}

// ---------------------------------------------------------------------------
// Kernel 2: h = X * W^T   via bf16x3 WMMA (fp32-accurate)
//   X  : [N, 128] fp32 row-major
//   W  : [128, 128] fp32 row-major (h[n][o] = sum_k X[n][k] * W[o][k])
//   h  : [N, 128] fp32
// Block = 256 threads = 8 waves; each wave -> 16 rows x 128 cols strip.
// Grid  = N / 128 = 1024 blocks.
// LDS: W staged as bf16 hi/lo planes, layout [o][k] (== global layout, since
// B fragment row n needs consecutive K — one 16B LDS read per 8-run).
// ---------------------------------------------------------------------------
__global__ void k_gemm_wmma(const float* __restrict__ X,
                            const float* __restrict__ W,
                            float* __restrict__ H) {
    __shared__ __bf16 sWhi[CC * CC];   // 32 KB
    __shared__ __bf16 sWlo[CC * CC];   // 32 KB

    const int tid = threadIdx.x;

    // --- cooperative stage of W into LDS as hi/lo bf16 (float4 coalesced) ---
    const float4* W4 = (const float4*)W;
    for (int i = tid; i < (CC * CC) / 4; i += 256) {
        float4 w = W4[i];
        int b = i * 4;
        bf_split(w.x, sWhi[b + 0], sWlo[b + 0]);
        bf_split(w.y, sWhi[b + 1], sWlo[b + 1]);
        bf_split(w.z, sWhi[b + 2], sWlo[b + 2]);
        bf_split(w.w, sWhi[b + 3], sWlo[b + 3]);
    }
    __syncthreads();

    const int wid     = tid >> 5;          // wave id 0..7
    const int lane    = tid & 31;
    const int laneRow = lane & 15;         // M (for A) / N (for B) within tile
    const int g       = lane >> 4;         // lane-group: selects K half-runs
    const int rowBase = blockIdx.x * 128 + wid * 16;

    v8f acc[8];
#pragma unroll
    for (int t = 0; t < 8; ++t) acc[t] = (v8f){0,0,0,0,0,0,0,0};

    const float* xr = X + (size_t)(rowBase + laneRow) * CC;

#pragma unroll
    for (int ks = 0; ks < 4; ++ks) {
        const int k0 = ks * 32;
        // ---- A fragment: lane holds K = {k0+8g .. +7} and {k0+16+8g .. +7} ----
        const float* a0 = xr + k0 + g * 8;
        float4 f0 = *(const float4*)(a0);
        float4 f1 = *(const float4*)(a0 + 4);
        float4 f2 = *(const float4*)(a0 + 16);
        float4 f3 = *(const float4*)(a0 + 20);
        float av[16] = { f0.x, f0.y, f0.z, f0.w, f1.x, f1.y, f1.z, f1.w,
                         f2.x, f2.y, f2.z, f2.w, f3.x, f3.y, f3.z, f3.w };
        v16bf a_hi, a_lo;
#pragma unroll
        for (int j = 0; j < 16; ++j) {
            __bf16 hi, lo;
            bf_split(av[j], hi, lo);
            a_hi[j] = hi; a_lo[j] = lo;
        }

#pragma unroll
        for (int t = 0; t < 8; ++t) {
            // ---- B fragment for output tile t (N = t*16 + laneRow) ----
            const int n = t * 16 + laneRow;
            const __bf16* bh = &sWhi[n * CC + k0 + g * 8];
            const __bf16* bl = &sWlo[n * CC + k0 + g * 8];
            bf16x8 bh0 = *(const bf16x8*)(bh);
            bf16x8 bh1 = *(const bf16x8*)(bh + 16);
            bf16x8 bl0 = *(const bf16x8*)(bl);
            bf16x8 bl1 = *(const bf16x8*)(bl + 16);
            v16bf b_hi, b_lo;
#pragma unroll
            for (int j = 0; j < 8; ++j) {
                b_hi[j] = bh0[j]; b_hi[8 + j] = bh1[j];
                b_lo[j] = bl0[j]; b_lo[8 + j] = bl1[j];
            }
            // bf16x3: hi*hi + hi*lo + lo*hi  (fp32 accumulate)
            acc[t] = __builtin_amdgcn_wmma_f32_16x16x32_bf16(
                false, a_hi, false, b_hi, (short)0, acc[t], false, false);
            acc[t] = __builtin_amdgcn_wmma_f32_16x16x32_bf16(
                false, a_hi, false, b_lo, (short)0, acc[t], false, false);
            acc[t] = __builtin_amdgcn_wmma_f32_16x16x32_bf16(
                false, a_lo, false, b_hi, (short)0, acc[t], false, false);
        }
    }

    // ---- store D: VGPR r, lanes 0-15 -> M=r ; lanes 16-31 -> M=r+8 ----
#pragma unroll
    for (int t = 0; t < 8; ++t) {
#pragma unroll
        for (int r = 0; r < 8; ++r) {
            int m = r + g * 8;
            H[(size_t)(rowBase + m) * CC + t * 16 + laneRow] = acc[t][r];
        }
    }
}

// ---------------------------------------------------------------------------
// Kernel 3: degree accumulation  deg[col[e]] += 1   (L2-resident atomics)
// ---------------------------------------------------------------------------
__global__ void k_degree(const int* __restrict__ ei, float* __restrict__ deg) {
    int e = blockIdx.x * blockDim.x + threadIdx.x;
    if (e < EE) atomicAdd(&deg[ei[EE + e]], 1.0f);
}

// ---------------------------------------------------------------------------
// Kernel 4: dis[i] = rsqrt(deg[i] + 1)   (+1 = self-loop), in place
// ---------------------------------------------------------------------------
__global__ void k_rsqrt(float* __restrict__ d) {
    int i = blockIdx.x * blockDim.x + threadIdx.x;
    if (i < NN) d[i] = rsqrtf(d[i] + 1.0f);
}

// ---------------------------------------------------------------------------
// Kernel 5: edge scatter.  One wave per edge; lane handles 4 channels.
// agg[col] += h[row] * (dis[row]*dis[col]).  agg (64MB) + h (64MB) fit in
// the 192MB L2, so the atomic adds never leave the chip.
// ---------------------------------------------------------------------------
__global__ void k_scatter(const float* __restrict__ h,
                          const int* __restrict__ ei,
                          const float* __restrict__ dis,
                          float* __restrict__ agg) {
    int e = blockIdx.x * (blockDim.x >> 5) + (threadIdx.x >> 5);
    if (e >= EE) return;
    int lane = threadIdx.x & 31;
    int r = ei[e];
    int c = ei[EE + e];
    float nrm = dis[r] * dis[c];
    float4 hv = ((const float4*)(h + (size_t)r * CC))[lane];
    float* dst = agg + (size_t)c * CC + lane * 4;
    atomicAdd(dst + 0, hv.x * nrm);
    atomicAdd(dst + 1, hv.y * nrm);
    atomicAdd(dst + 2, hv.z * nrm);
    atomicAdd(dst + 3, hv.w * nrm);
}

// ---------------------------------------------------------------------------
// Kernel 6: fused finalize + residual + 16-node sum-pool + 3-layer MLP.
// One 128-thread block per graph.
// conv_out = relu(agg + h*dis^2 (self loop) + conv_b); x = conv_out + state;
// p = sum over 16 nodes; y1 = relu(W1 p + b1); y2 = relu(W2 y1 + b2);
// out = W3 y2 + b3.
// ---------------------------------------------------------------------------
__global__ void k_pool_mlp(const float* __restrict__ h,
                           const float* __restrict__ agg,
                           const float* __restrict__ dis,
                           const float* __restrict__ state,
                           const float* __restrict__ conv_b,
                           const float* __restrict__ W1, const float* __restrict__ b1,
                           const float* __restrict__ W2, const float* __restrict__ b2,
                           const float* __restrict__ W3, const float* __restrict__ b3,
                           float* __restrict__ out) {
    __shared__ float p[CC];
    __shared__ float y1[64];
    __shared__ float red[64];

    const int b = blockIdx.x;
    const int c = threadIdx.x;

    float bias = conv_b[c];
    float s = 0.0f;
    const int base = b * ACTD;
#pragma unroll
    for (int j = 0; j < ACTD; ++j) {
        int i = base + j;
        float d = dis[i];
        size_t off = (size_t)i * CC + c;
        float v = agg[off] + h[off] * (d * d) + bias;   // conv + self-loop + bias
        v = fmaxf(v, 0.0f);                             // relu
        s += v + state[off];                            // residual, pooled
    }
    p[c] = s;
    __syncthreads();

    if (c < 64) {
        float a = b1[c];
#pragma unroll 4
        for (int k = 0; k < CC; ++k) a += W1[c * CC + k] * p[k];
        y1[c] = fmaxf(a, 0.0f);
    }
    __syncthreads();

    if (c < 64) {
        float a = b2[c];
#pragma unroll 4
        for (int k = 0; k < 64; ++k) a += W2[c * 64 + k] * y1[k];
        red[c] = W3[c] * fmaxf(a, 0.0f);
    }
    __syncthreads();

    if (c == 0) {
        float a = b3[0];
#pragma unroll
        for (int k = 0; k < 64; ++k) a += red[k];
        out[b] = a;
    }
}

// ---------------------------------------------------------------------------
// Host launcher
// ---------------------------------------------------------------------------
extern "C" void kernel_launch(void* const* d_in, const int* in_sizes, int n_in,
                              void* d_out, int out_size, void* d_ws, size_t ws_size,
                              hipStream_t stream) {
    const float* state  = (const float*)d_in[0];
    const int*   ei     = (const int*)  d_in[1];   // [2,E]: row=ei[0..E), col=ei[E..2E)
    const float* conv_W = (const float*)d_in[2];
    const float* conv_b = (const float*)d_in[3];
    const float* W1     = (const float*)d_in[4];
    const float* b1     = (const float*)d_in[5];
    const float* W2     = (const float*)d_in[6];
    const float* b2     = (const float*)d_in[7];
    const float* W3     = (const float*)d_in[8];
    const float* b3     = (const float*)d_in[9];
    float*       out    = (float*)d_out;

    // Workspace layout: h [N*128] | agg [N*128] | deg/dis [N]
    float* h   = (float*)d_ws;
    float* agg = h   + (size_t)NN * CC;
    float* dis = agg + (size_t)NN * CC;

    // 1) zero agg + deg (contiguous N*129 floats)
    k_zero<<<2048, 256, 0, stream>>>(agg, (size_t)NN * (CC + 1));

    // 2) h = X W^T  (bf16x3 WMMA)
    k_gemm_wmma<<<NN / 128, 256, 0, stream>>>(state, conv_W, h);

    // 3) degree
    k_degree<<<EE / 256, 256, 0, stream>>>(ei, dis);

    // 4) dis = rsqrt(deg + 1)
    k_rsqrt<<<NN / 256, 256, 0, stream>>>(dis);

    // 5) edge scatter (L2-resident atomics)
    k_scatter<<<EE / 8, 256, 0, stream>>>(h, ei, dis, agg);

    // 6) fused finalize + pool + MLP
    k_pool_mlp<<<BB, 128, 0, stream>>>(h, agg, dis, state, conv_b,
                                       W1, b1, W2, b2, W3, b3, out);
}
